// SchNetInteraction_24953759989865
// MI455X (gfx1250) — matrix operand order repaired and checked by reading, not compiled, not run on hardware
//
#include <hip/hip_runtime.h>

typedef __attribute__((ext_vector_type(16))) __bf16 v16bf;
typedef __attribute__((ext_vector_type(8)))  float  v8f;

union FragBF { v16bf v; unsigned short h[16]; unsigned u[8]; uint4 q[2]; };

__device__ __forceinline__ unsigned short f32_bf16(float x) {
  unsigned u = __float_as_uint(x);
  unsigned r = u + 0x7FFFu + ((u >> 16) & 1u);   // round-to-nearest-even
  return (unsigned short)(r >> 16);
}

__device__ __forceinline__ unsigned pack2_bf16(float a, float b) {
  return (unsigned)f32_bf16(a) | ((unsigned)f32_bf16(b) << 16);
}

__device__ __forceinline__ float ssp(float x) {
  // softplus(x) - log(2), numerically stable
  float t = __expf(-fabsf(x));
  return fmaxf(x, 0.0f) + __logf(1.0f + t) - 0.6931471824645996f;
}

__device__ __forceinline__ v8f wmma_bf16(v16bf a, v16bf b, v8f c) {
  return __builtin_amdgcn_wmma_f32_16x16x32_bf16(false, a, false, b, (short)0, c,
                                                 false, false);
}

// A fragment (16x32) from row-major bf16 tile (ld=128): two 16B LDS loads per lane.
__device__ __forceinline__ v16bf load_a_frag(const unsigned short* tile, int k0, int lane) {
  int lane16 = lane & 15;
  int hk = (lane >> 4) ? 8 : 0;
  const unsigned short* base = tile + lane16 * 128 + k0 + hk;
  FragBF f;
  f.q[0] = *(const uint4*)(base);
  f.q[1] = *(const uint4*)(base + 16);
  return f.v;
}

// B fragment (32x16) from pre-transposed bf16 weights Wt[n][k] (ld=128): two 16B loads.
__device__ __forceinline__ v16bf load_b_frag(const unsigned short* __restrict__ Wt,
                                             int k0, int n0, int lane) {
  int lane16 = lane & 15;
  int hk = (lane >> 4) ? 8 : 0;
  const unsigned short* base = Wt + (n0 + lane16) * 128 + k0 + hk;
  FragBF f;
  f.q[0] = *(const uint4*)(base);
  f.q[1] = *(const uint4*)(base + 16);
  return f.v;
}

// Preload all 4 A fragments (one LDS clause), then run the 4-WMMA chain.
__device__ __forceinline__ v8f gemm_128(const unsigned short* tile, const v16bf* B,
                                        v8f acc, int lane) {
  v16bf A0 = load_a_frag(tile, 0,  lane);
  v16bf A1 = load_a_frag(tile, 32, lane);
  v16bf A2 = load_a_frag(tile, 64, lane);
  v16bf A3 = load_a_frag(tile, 96, lane);
  acc = wmma_bf16(A0, B[0], acc);
  acc = wmma_bf16(A1, B[1], acc);
  acc = wmma_bf16(A2, B[2], acc);
  acc = wmma_bf16(A3, B[3], acc);
  return acc;
}

__global__ void __launch_bounds__(256) k_zero_f32(float* __restrict__ p, long n) {
  long i = (long)blockIdx.x * 256 + threadIdx.x;
  long stride = (long)gridDim.x * 256;
  for (; i < n; i += stride) p[i] = 0.0f;
}

// Wt[n][k] = bf16(W[k][n]) for a 128x128 fp32 matrix. Grid: 64 blocks x 256.
__global__ void __launch_bounds__(256) k_wprep(const float* __restrict__ W,
                                               unsigned short* __restrict__ Wt) {
  int idx = blockIdx.x * 256 + threadIdx.x;   // 0..16383
  int k = idx >> 7, n = idx & 127;
  Wt[n * 128 + k] = f32_bf16(W[k * 128 + n]);
}

// Stage 16 fp32 rows (ld=128) -> bf16 LDS tile. Each thread: 8 consecutive elems.
__device__ __forceinline__ void stage_tile_bf16(const float* __restrict__ X, long rowBase,
                                                int nRows, unsigned short* sT, int tid) {
  int b = tid * 8;                 // 0..2047
  long r = rowBase + (b >> 7);
  if (r >= nRows) r = nRows - 1;   // only taken in tail tiles
  int c = b & 127;
  const float4 f0 = *(const float4*)(X + r * 128 + c);
  const float4 f1 = *(const float4*)(X + r * 128 + c + 4);
  uint4 pk;
  pk.x = pack2_bf16(f0.x, f0.y);
  pk.y = pack2_bf16(f0.z, f0.w);
  pk.z = pack2_bf16(f1.x, f1.y);
  pk.w = pack2_bf16(f1.z, f1.w);
  *(uint4*)(sT + b) = pk;
}

#define TPB_NODE 2   // 16-row tiles per block, node-side kernels
#define TPB_EDGE 8   // 16-edge tiles per block, edge kernel

// Y = X @ W (no bias, no activation).
__global__ void __launch_bounds__(256) k_rowgemm(const float* __restrict__ X,
                                                 const unsigned short* __restrict__ WnT,
                                                 float* __restrict__ Y, int nRows) {
  __shared__ unsigned short sX[16 * 128];
  int tid = threadIdx.x, lane = tid & 31, wave = tid >> 5;
  int n0 = wave * 16, lane16 = lane & 15, hi = (lane >> 4) * 8;
  int col = n0 + lane16;

  v16bf B[4];
#pragma unroll
  for (int kt = 0; kt < 4; ++kt) B[kt] = load_b_frag(WnT, kt * 32, n0, lane);

  for (int t = 0; t < TPB_NODE; ++t) {
    int r0 = (blockIdx.x * TPB_NODE + t) * 16;
    if (r0 >= nRows) break;                         // uniform
    stage_tile_bf16(X, r0, nRows, sX, tid);
    __syncthreads();

    v8f acc = {};
    acc = gemm_128(sX, B, acc, lane);

    if (r0 + 16 <= nRows) {                         // uniform fast path
      float* p = Y + (long)(r0 + hi) * 128 + col;
#pragma unroll
      for (int vr = 0; vr < 8; ++vr) p[vr * 128] = acc[vr];
    } else {
#pragma unroll
      for (int vr = 0; vr < 8; ++vr) {
        int r = r0 + vr + hi;
        if (r < nRows) Y[(long)r * 128 + col] = acc[vr];
      }
    }
    __syncthreads();                                // before next stage overwrites sX
  }
}

// Fused edge pipeline: f = ssp(edge@We1+be1)@We2+be2 ; msg = xn[src]*f ; agg[dst] += msg
__global__ void __launch_bounds__(256) k_edge(const float* __restrict__ edge,
                                              const int* __restrict__ eidx,
                                              const unsigned short* __restrict__ We1T,
                                              const float* __restrict__ be1,
                                              const unsigned short* __restrict__ We2T,
                                              const float* __restrict__ be2,
                                              const float* __restrict__ xn,
                                              float* __restrict__ agg, int Mtot) {
  __shared__ unsigned short sE[16 * 128];
  __shared__ unsigned short sH[16 * 128];
  __shared__ int sDst[16], sSrc[16];
  int tid = threadIdx.x, lane = tid & 31, wave = tid >> 5;
  int n0 = wave * 16, lane16 = lane & 15, hi = (lane >> 4) * 8;
  int col = n0 + lane16;

  v16bf B1[4], B2[4];
#pragma unroll
  for (int kt = 0; kt < 4; ++kt) {
    B1[kt] = load_b_frag(We1T, kt * 32, n0, lane);
    B2[kt] = load_b_frag(We2T, kt * 32, n0, lane);
  }
  float bias1 = be1[col];
  float bias2 = be2[col];

  for (int t = 0; t < TPB_EDGE; ++t) {
    long e0 = ((long)blockIdx.x * TPB_EDGE + t) * 16;
    if (e0 >= Mtot) break;                          // uniform
    stage_tile_bf16(edge, e0, Mtot, sE, tid);
    if (tid < 16) {
      long e = e0 + tid;
      if (e >= Mtot) e = Mtot - 1;
      sDst[tid] = eidx[e];                          // edge_index[0] = receiving
      sSrc[tid] = eidx[Mtot + e];                   // edge_index[1] = outgoing
    }
    __syncthreads();

    // Issue the xn gather early: its L2 latency hides under both GEMM layers.
    float xv[8];
    int   dIdx[8];
#pragma unroll
    for (int vr = 0; vr < 8; ++vr) {
      int er = vr + hi;
      dIdx[vr] = sDst[er];
      xv[vr]   = xn[(long)sSrc[er] * 128 + col];
    }

    // dense1 + ssp
    v8f acc;
#pragma unroll
    for (int vr = 0; vr < 8; ++vr) acc[vr] = bias1;
    acc = gemm_128(sE, B1, acc, lane);
#pragma unroll
    for (int vr = 0; vr < 8; ++vr)
      sH[(vr + hi) * 128 + col] = f32_bf16(ssp(acc[vr]));
    __syncthreads();

    // dense2 (linear)
    v8f acc2;
#pragma unroll
    for (int vr = 0; vr < 8; ++vr) acc2[vr] = bias2;
    acc2 = gemm_128(sH, B2, acc2, lane);

    // scatter-add msg = xv * f into agg[dst] (L2-resident)
    if (e0 + 16 <= Mtot) {                          // uniform fast path
#pragma unroll
      for (int vr = 0; vr < 8; ++vr)
        atomicAdd(&agg[(long)dIdx[vr] * 128 + col], acc2[vr] * xv[vr]);
    } else {
#pragma unroll
      for (int vr = 0; vr < 8; ++vr) {
        if (e0 + vr + hi < Mtot)
          atomicAdd(&agg[(long)dIdx[vr] * 128 + col], acc2[vr] * xv[vr]);
      }
    }
    __syncthreads();                                // before next stage overwrites sE/sIdx
  }
}

// out = node + ssp(agg@W2+b2)@W3 + b3
__global__ void __launch_bounds__(256) k_out(const float* __restrict__ agg,
                                             const unsigned short* __restrict__ W2T,
                                             const float* __restrict__ b2,
                                             const unsigned short* __restrict__ W3T,
                                             const float* __restrict__ b3,
                                             const float* __restrict__ node,
                                             float* __restrict__ out, int nRows) {
  __shared__ unsigned short sA[16 * 128];
  __shared__ unsigned short sH[16 * 128];
  int tid = threadIdx.x, lane = tid & 31, wave = tid >> 5;
  int n0 = wave * 16, lane16 = lane & 15, hi = (lane >> 4) * 8;
  int col = n0 + lane16;

  v16bf B2[4], B3[4];
#pragma unroll
  for (int kt = 0; kt < 4; ++kt) {
    B2[kt] = load_b_frag(W2T, kt * 32, n0, lane);
    B3[kt] = load_b_frag(W3T, kt * 32, n0, lane);
  }
  float bias2 = b2[col];
  float bias3 = b3[col];

  for (int t = 0; t < TPB_NODE; ++t) {
    int r0 = (blockIdx.x * TPB_NODE + t) * 16;
    if (r0 >= nRows) break;                         // uniform
    stage_tile_bf16(agg, r0, nRows, sA, tid);
    __syncthreads();

    // residual load issued early; hides under both GEMMs
    float nv[8];
    bool full = (r0 + 16 <= nRows);
    if (full) {
      const float* np = node + (long)(r0 + hi) * 128 + col;
#pragma unroll
      for (int vr = 0; vr < 8; ++vr) nv[vr] = np[vr * 128];
    } else {
#pragma unroll
      for (int vr = 0; vr < 8; ++vr) {
        int r = r0 + vr + hi;
        nv[vr] = (r < nRows) ? node[(long)r * 128 + col] : 0.0f;
      }
    }

    v8f acc;
#pragma unroll
    for (int vr = 0; vr < 8; ++vr) acc[vr] = bias2;
    acc = gemm_128(sA, B2, acc, lane);
#pragma unroll
    for (int vr = 0; vr < 8; ++vr)
      sH[(vr + hi) * 128 + col] = f32_bf16(ssp(acc[vr]));
    __syncthreads();

    v8f acc2;
#pragma unroll
    for (int vr = 0; vr < 8; ++vr) acc2[vr] = bias3;
    acc2 = gemm_128(sH, B3, acc2, lane);

    if (full) {                                     // uniform fast path
      float* op = out + (long)(r0 + hi) * 128 + col;
#pragma unroll
      for (int vr = 0; vr < 8; ++vr) op[vr * 128] = nv[vr] + acc2[vr];
    } else {
#pragma unroll
      for (int vr = 0; vr < 8; ++vr) {
        int r = r0 + vr + hi;
        if (r < nRows) out[(long)r * 128 + col] = nv[vr] + acc2[vr];
      }
    }
    __syncthreads();                                // before next stage overwrites sA
  }
}

extern "C" void kernel_launch(void* const* d_in, const int* in_sizes, int n_in,
                              void* d_out, int out_size, void* d_ws, size_t ws_size,
                              hipStream_t stream) {
  (void)n_in; (void)out_size; (void)ws_size;
  const float* node = (const float*)d_in[0];
  const float* edge = (const float*)d_in[1];
  const int*   eidx = (const int*)d_in[2];
  const float* Wn   = (const float*)d_in[3];
  const float* We1  = (const float*)d_in[4];
  const float* be1  = (const float*)d_in[5];
  const float* We2  = (const float*)d_in[6];
  const float* be2  = (const float*)d_in[7];
  const float* W2   = (const float*)d_in[8];
  const float* b2   = (const float*)d_in[9];
  const float* W3   = (const float*)d_in[10];
  const float* b3   = (const float*)d_in[11];

  const int N = in_sizes[0] / 128;
  const int M = in_sizes[1] / 128;

  float* xn  = (float*)d_ws;                         // [N,128] fp32
  float* agg = xn + (size_t)N * 128;                 // [N,128] fp32
  unsigned short* WnT  = (unsigned short*)(agg + (size_t)N * 128);
  unsigned short* We1T = WnT  + 128 * 128;           // bf16 transposed weights
  unsigned short* We2T = We1T + 128 * 128;
  unsigned short* W2T  = We2T + 128 * 128;
  unsigned short* W3T  = W2T  + 128 * 128;

  const int nodeTiles = (N + 15) / 16;
  const int edgeTiles = (M + 15) / 16;
  const int nodeBlocks = (nodeTiles + TPB_NODE - 1) / TPB_NODE;
  const int edgeBlocks = (edgeTiles + TPB_EDGE - 1) / TPB_EDGE;

  k_zero_f32<<<1024, 256, 0, stream>>>(agg, (long)N * 128);
  k_wprep<<<64, 256, 0, stream>>>(Wn,  WnT);
  k_wprep<<<64, 256, 0, stream>>>(We1, We1T);
  k_wprep<<<64, 256, 0, stream>>>(We2, We2T);
  k_wprep<<<64, 256, 0, stream>>>(W2,  W2T);
  k_wprep<<<64, 256, 0, stream>>>(W3,  W3T);
  k_rowgemm<<<nodeBlocks, 256, 0, stream>>>(node, WnT, xn, N);
  k_edge<<<edgeBlocks, 256, 0, stream>>>(edge, eidx, We1T, be1, We2T, be2, xn, agg, M);
  k_out<<<nodeBlocks, 256, 0, stream>>>(agg, W2T, b2, W3T, b3, node, (float*)d_out, N);
}
